// Tree_Net_51797305590068
// MI455X (gfx1250) — matrix operand order repaired and checked by reading, not compiled, not run on hardware
//
#include <hip/hip_runtime.h>
#include <hip/hip_bf16.h>

// ---------------------------------------------------------------------------
// Types for CDNA5 WMMA (wave32): v_wmma_f32_16x16x32_bf16
// ---------------------------------------------------------------------------
typedef __bf16 bf16;
typedef __attribute__((ext_vector_type(16))) __bf16 bf16x16;
typedef __attribute__((ext_vector_type(8)))  __bf16 bf16x8;
typedef __attribute__((ext_vector_type(8)))  float  f32x8;

// ---------------------------------------------------------------------------
// CDNA5 async global->LDS copy (ASYNCcnt-tracked, no VGPR round trip).
// Guarded: falls back to load+ds_store if the builtins are unavailable.
// The builtin signature wants v4i pointers: AS1 (global) src, AS3 (LDS) dst.
// ---------------------------------------------------------------------------
#if defined(__has_builtin)
#if __has_builtin(__builtin_amdgcn_global_load_async_to_lds_b128) && \
    __has_builtin(__builtin_amdgcn_s_wait_asynccnt)
#define HAVE_ASYNC_LDS 1
#endif
#endif

typedef int v4i __attribute__((ext_vector_type(4)));
#if defined(HAVE_ASYNC_LDS)
typedef __attribute__((address_space(1))) v4i as1_v4i;
typedef __attribute__((address_space(3))) v4i as3_v4i;
#endif

// Copy 16B from global to LDS for this lane.
__device__ __forceinline__ void cp16_g2l(const void* gp, void* lp) {
#if defined(HAVE_ASYNC_LDS)
  __builtin_amdgcn_global_load_async_to_lds_b128(
      (as1_v4i*)(unsigned long long)gp,
      (as3_v4i*)(unsigned)(unsigned long long)lp, 0, 0);
#else
  *(uint4*)lp = *(const uint4*)gp;
#endif
}
// Wait for this wave's outstanding async copies (no-op in fallback; the
// implicit loadcnt/dscnt waits cover the synchronous path).
__device__ __forceinline__ void cp_commit() {
#if defined(HAVE_ASYNC_LDS)
  __builtin_amdgcn_s_wait_asynccnt(0);
#endif
}

__device__ __forceinline__ f32x8 zero8() {
  f32x8 z;
#pragma unroll
  for (int i = 0; i < 8; ++i) z[i] = 0.0f;
  return z;
}

// B operand: 32x16 (KxN) bf16 tile from row-major [rows][ld] buffer where a
// "row" is an N index and columns are K. Lane 0-15 -> N=row+lane, K=kk..kk+15;
// lane 16-31 -> N=row+lane-16, K=kk+16..kk+31. (contiguous k order, same as frag_a)
__device__ __forceinline__ bf16x16 frag_b(const bf16* p, int row, int ld, int kk) {
  int lane = threadIdx.x & 31;
  const bf16* q = p + (size_t)(row + (lane & 15)) * ld + kk + ((lane < 16) ? 0 : 16);
  return *(const bf16x16*)q;
}

// A operand: 16x32 (MxK) bf16 tile. Lane 0-15 -> M=row+lane, K=kk..kk+15;
// lane 16-31 -> same M rows, K=kk+16..kk+31. (contiguous k order, same as frag_b)
__device__ __forceinline__ bf16x16 frag_a(const bf16* p, int row, int ld, int kk) {
  int lane = threadIdx.x & 31;
  const bf16* q = p + (size_t)(row + (lane & 15)) * ld + kk + ((lane < 16) ? 0 : 16);
  bf16x8 lo = *(const bf16x8*)q;
  bf16x8 hi = *(const bf16x8*)(q + 8);
  bf16x16 r;
#pragma unroll
  for (int i = 0; i < 8; ++i) { r[i] = lo[i]; r[i + 8] = hi[i]; }
  return r;
}

__device__ __forceinline__ f32x8 wmma_bf16(bf16x16 a, bf16x16 b, f32x8 c) {
  return __builtin_amdgcn_wmma_f32_16x16x32_bf16(false, a, false, b, (short)0, c,
                                                 false, false);
}

// ---------------------------------------------------------------------------
// Block-wide sum over 256 threads (8 wave32s)
// ---------------------------------------------------------------------------
__device__ __forceinline__ float blockReduceSum(float v, float* scr) {
  int lane = threadIdx.x & 31, wid = threadIdx.x >> 5;
#pragma unroll
  for (int o = 16; o > 0; o >>= 1) v += __shfl_xor(v, o);
  if (lane == 0) scr[wid] = v;
  __syncthreads();
  float tot = 0.0f;
#pragma unroll
  for (int i = 0; i < 8; ++i) tot += scr[i];
  __syncthreads();
  return tot;
}

// ---------------------------------------------------------------------------
// Agent-scope ticket grid barrier (64 workgroups). Acquire/release atomics make
// the compiler emit global_wb / global_inv so cross-WGP data is coherent.
// ---------------------------------------------------------------------------
__device__ __forceinline__ void grid_barrier(unsigned* cnt) {
  __threadfence();
  __syncthreads();
  if (threadIdx.x == 0) {
    unsigned t = __hip_atomic_fetch_add(cnt, 1u, __ATOMIC_ACQ_REL,
                                        __HIP_MEMORY_SCOPE_AGENT);
    unsigned target = t - (t % 64u) + 64u;
    while (__hip_atomic_load(cnt, __ATOMIC_ACQUIRE, __HIP_MEMORY_SCOPE_AGENT) <
           target)
      __builtin_amdgcn_s_sleep(1);
  }
  __syncthreads();
  __threadfence();
}

__device__ __forceinline__ float sigf(float x) { return 1.0f / (1.0f + __expf(-x)); }

// ---------------------------------------------------------------------------
// f32 -> bf16 packing. One block per destination row; rows >= src_rows -> 0.
// ---------------------------------------------------------------------------
__global__ __launch_bounds__(256)
void k_pack(const float* __restrict__ src, bf16* __restrict__ dst,
            int cols, int src_ld, int dst_ld, int src_rows) {
  int r = blockIdx.x;
  for (int c = threadIdx.x; c < cols; c += blockDim.x) {
    float v = (r < src_rows) ? src[(size_t)r * src_ld + c] : 0.0f;
    dst[(size_t)r * dst_ld + c] = (bf16)v;
  }
}

// Reverse along sequence (per batch of 128) while converting to bf16.
__global__ __launch_bounds__(256)
void k_pack_rev(const float* __restrict__ src, bf16* __restrict__ dst, int cols) {
  int row = blockIdx.x;                 // b*128 + s
  int b = row >> 7, s = row & 127;
  int srow = b * 128 + (127 - s);
  for (int c = threadIdx.x; c < cols; c += blockDim.x)
    dst[(size_t)row * cols + c] = (bf16)src[(size_t)srow * cols + c];
}

__global__ void k_zero_u32(unsigned* p, int n) {
  int i = blockIdx.x * blockDim.x + threadIdx.x;
  if (i < n) p[i] = 0u;
}

// ---------------------------------------------------------------------------
// Generic bf16 WMMA GEMM:  C[m,n] = sum_k A[m,k] * W[n,k] (+bias, +leaky-relu)
// Block tile 64x128, BK=32, 256 threads = 8 waves, each wave: 16 N-cols x 64 M.
// LDS double-buffered with async global->LDS copies: tile k+1 streams in while
// the 4 WMMAs of tile k execute; s_wait_asynccnt + barrier per K-step.
// grid = (Npad/128, M/64). Stores only cols < n_real (handles N=511/152).
// ---------------------------------------------------------------------------
__global__ __launch_bounds__(256)
void k_gemm(const bf16* __restrict__ A, int lda,
            const bf16* __restrict__ W, int ldw,
            float* __restrict__ C, int ldc,
            const float* __restrict__ bias,
            int K, int n_real, int act) {
  __shared__ bf16 sA[2][64 * 32];
  __shared__ bf16 sB[2][128 * 32];
  int tid = threadIdx.x;
  int wv = tid >> 5, lane = tid & 31;
  int m0 = blockIdx.y * 64;
  int n0 = blockIdx.x * 128;
  f32x8 acc[4];
#pragma unroll
  for (int i = 0; i < 4; ++i) acc[i] = zero8();

  int row = tid >> 2, ch = (tid & 3) * 8;
  const bf16* pa  = A + (size_t)(m0 + row) * lda + ch;
  const bf16* pb0 = W + (size_t)(n0 + row) * ldw + ch;
  const bf16* pb1 = W + (size_t)(n0 + row + 64) * ldw + ch;
  size_t lo = (size_t)row * 32 + ch;

  // Preload K-slab 0 into buffer 0.
  cp16_g2l(pa, sA[0] + lo);
  cp16_g2l(pb0, sB[0] + lo);
  cp16_g2l(pb1, sB[0] + (size_t)64 * 32 + lo);
  cp_commit();
  __syncthreads();

  for (int kk = 0; kk < K; kk += 32) {
    int cur = (kk >> 5) & 1, nxt = cur ^ 1;
    if (kk + 32 < K) {  // stream next slab while computing this one
      cp16_g2l(pa + kk + 32, sA[nxt] + lo);
      cp16_g2l(pb0 + kk + 32, sB[nxt] + lo);
      cp16_g2l(pb1 + kk + 32, sB[nxt] + (size_t)64 * 32 + lo);
    }
    if (kk + 64 < K) {  // global_prefetch_b8, WGP scope
      __builtin_prefetch(pa + kk + 64, 0, 3);
      __builtin_prefetch(pb0 + kk + 64, 0, 3);
    }
    bf16x16 bfrag = frag_b(sB[cur], wv * 16, 32, 0);
#pragma unroll
    for (int mt = 0; mt < 4; ++mt) {
      bf16x16 afrag = frag_a(sA[cur], mt * 16, 32, 0);
      acc[mt] = wmma_bf16(afrag, bfrag, acc[mt]);
    }
    cp_commit();
    __syncthreads();
  }

  int col = n0 + wv * 16 + (lane & 15);
  if (col < n_real) {
    float bv = bias ? bias[col] : 0.0f;
#pragma unroll
    for (int mt = 0; mt < 4; ++mt) {
      int rbase = m0 + mt * 16 + ((lane < 16) ? 0 : 8);
#pragma unroll
      for (int r = 0; r < 8; ++r) {
        float v = acc[mt][r] + bv;
        if (act) v = v > 0.0f ? v : 0.01f * v;
        C[(size_t)(rbase + r) * ldc + col] = v;
      }
    }
  }
}

// ---------------------------------------------------------------------------
// Persistent bidirectional LSTM scan. 64 WGs: dir = wg/32, n = wg%32 owns gate
// columns [64n, 64n+64). w_hh slice (64x512 bf16 = 64KB) is LDS-resident for
// all 128 steps; h (64x512 bf16) is async-staged into LDS each step.
// Per step: 64x64 gate tile = h @ w_hh_slice^T via WMMA, + xp, -> gbuf;
// grid barrier; WGs n<8 do the sigmoid/tanh pointwise update of h/c and emit
// hs into the concatenated [fwd|bwd] bf16 activation buffer; grid barrier.
// ---------------------------------------------------------------------------
__global__ __launch_bounds__(256)
void k_lstm(const float* __restrict__ xp_f, const float* __restrict__ xp_b,
            const bf16* __restrict__ whh_f, const bf16* __restrict__ whh_b,
            bf16* __restrict__ h_state, float* __restrict__ c_state,
            float* __restrict__ gbuf, bf16* __restrict__ hs_cat,
            unsigned* __restrict__ cnt) {
  extern __shared__ char lstm_smem[];
  bf16* wl = (bf16*)lstm_smem;      // [64][512] resident w_hh slice
  bf16* hl = wl + 64 * 512;         // [64][512] staged h

  int wg = blockIdx.x;
  int d = wg >> 5, n = wg & 31;
  int tid = threadIdx.x, wv = tid >> 5, lane = tid & 31;

  const float* xp = d ? xp_b : xp_f;
  const bf16* whh = d ? whh_b : whh_f;
  bf16* h = h_state + (size_t)d * 64 * 512;
  float* c = c_state + (size_t)d * 64 * 512;
  float* g = gbuf + (size_t)d * 64 * 2048;

  {  // one-time async load of this WG's 64 contiguous w_hh rows into LDS
    const bf16* src = whh + (size_t)n * 64 * 512;
#pragma unroll
    for (int i = 0; i < 16; ++i) {
      size_t e = ((size_t)tid + i * 256) * 8;  // bf16 elements (16B chunks)
      cp16_g2l(src + e, wl + e);
    }
    cp_commit();
  }
  if (n < 8) {  // zero-init h0 / c0 (each of WGs 0..7 covers 1/8)
    for (int e = tid; e < 4096; e += 256) {
      int off = n * 4096 + e;
      h[off] = (bf16)0.0f;
      c[off] = 0.0f;
    }
  }
  grid_barrier(cnt);

  int m2 = wv >> 2;            // 0..1 -> batch rows [32*m2, 32*m2+32)
  int n16 = (wv & 3) * 16;     // gate-col sub-tile within this WG's 64

  for (int t = 0; t < 128; ++t) {
    {  // async-stage full h (64x512 bf16) into LDS
#pragma unroll
      for (int i = 0; i < 16; ++i) {
        size_t e = ((size_t)tid + i * 256) * 8;
        cp16_g2l(h + e, hl + e);
      }
      cp_commit();
    }
    __syncthreads();

    f32x8 acc0 = zero8(), acc1 = zero8();
#pragma unroll 4
    for (int kk = 0; kk < 512; kk += 32) {
      bf16x16 bfrag = frag_b(wl, n16, 512, kk);
      bf16x16 a0 = frag_a(hl, m2 * 32, 512, kk);
      bf16x16 a1 = frag_a(hl, m2 * 32 + 16, 512, kk);
      acc0 = wmma_bf16(a0, bfrag, acc0);
      acc1 = wmma_bf16(a1, bfrag, acc1);
    }

    int gcol = n * 64 + n16 + (lane & 15);
#pragma unroll
    for (int half = 0; half < 2; ++half) {
      f32x8 a = half ? acc1 : acc0;
      int rbase = m2 * 32 + half * 16 + ((lane < 16) ? 0 : 8);
#pragma unroll
      for (int r = 0; r < 8; ++r) {
        int b_ = rbase + r;
        g[(size_t)b_ * 2048 + gcol] =
            a[r] + xp[((size_t)b_ * 128 + t) * 2048 + gcol];
      }
    }
    grid_barrier(cnt);

    if (n < 8) {  // pointwise gate update for hidden cols [64n, 64n+64)
      for (int e = tid; e < 4096; e += 256) {
        int b_ = e >> 6, j = n * 64 + (e & 63);
        float ig = g[(size_t)b_ * 2048 + j];
        float fg = g[(size_t)b_ * 2048 + 512 + j];
        float gg = g[(size_t)b_ * 2048 + 1024 + j];
        float og = g[(size_t)b_ * 2048 + 1536 + j];
        float cn = sigf(fg) * c[b_ * 512 + j] + sigf(ig) * tanhf(gg);
        float hn = sigf(og) * tanhf(cn);
        c[b_ * 512 + j] = cn;
        h[b_ * 512 + j] = (bf16)hn;
        int s_out = d ? (127 - t) : t;  // bwd scan writes reversed position
        hs_cat[((size_t)b_ * 128 + s_out) * 1024 + d * 512 + j] = (bf16)hn;
      }
    }
    grid_barrier(cnt);
  }
}

// ---------------------------------------------------------------------------
// Row L2-normalize `combined` with original_pos scatter; emit f32 + bf16.
// One block (256 thr) per row of 512.
// ---------------------------------------------------------------------------
__global__ __launch_bounds__(256)
void k_rownorm(const float* __restrict__ comb, const int* __restrict__ pos,
               float* __restrict__ wv, bf16* __restrict__ wvb) {
  __shared__ float scr[8];
  int row = blockIdx.x;  // b*128 + s
  int b = row >> 7;
  int tgt = pos[row * 2 + 0];
  int src = pos[row * 2 + 1];
  const float* x = comb + ((size_t)b * 128 + src) * 512;
  int tid = threadIdx.x;
  float v0 = x[tid], v1 = x[tid + 256];
  float ss = blockReduceSum(v0 * v0 + v1 * v1, scr);
  float inv = 1.0f / fmaxf(sqrtf(ss), 1e-12f);
  size_t o = ((size_t)b * 128 + tgt) * 512;
  wv[o + tid] = v0 * inv;
  wv[o + tid + 256] = v1 * inv;
  wvb[o + tid] = (bf16)(v0 * inv);
  wvb[o + tid + 256] = (bf16)(v1 * inv);
}

// ---------------------------------------------------------------------------
// Tree composition: one WG per batch item runs all 127 sequential steps with
// the phrase table (127x512 f32 = 260KB) fully resident in CDNA5's large LDS.
// circular correlation c[k] = sum_j a[j] * b[(j+k) mod 512], all l2-normalized.
// Flushes phrase vectors to bf16 for the final WMMA projection.
// ---------------------------------------------------------------------------
__global__ __launch_bounds__(256)
void k_compose(const float* __restrict__ wordvec, const int* __restrict__ comp,
               bf16* __restrict__ phrase_bf) {
  extern __shared__ float ph[];  // 127*512 floats
  __shared__ float av[512], bv[512], araw0[512];
  __shared__ float scr[8];
  int b = blockIdx.x, tid = threadIdx.x;

  for (int i = tid; i < 127 * 512; i += 256) ph[i] = 0.0f;
  __syncthreads();

  for (int step = 0; step < 127; ++step) {
    const int* info = comp + ((size_t)b * 127 + step) * 4;
    int t = info[0], p = info[1], l = info[2], r = info[3];
    const float* lp = (l < 128) ? (wordvec + ((size_t)b * 128 + l) * 512)
                                : (ph + (size_t)(l - 128) * 512);
    const float* rp = (r < 128) ? (wordvec + ((size_t)b * 128 + r) * 512)
                                : (ph + (size_t)(r - 128) * 512);
    float l0 = lp[tid], l1 = lp[tid + 256];
    float r0 = rp[tid], r1 = rp[tid + 256];
    araw0[tid] = l0;
    araw0[tid + 256] = l1;
    float lss = blockReduceSum(l0 * l0 + l1 * l1, scr);
    float rss = blockReduceSum(r0 * r0 + r1 * r1, scr);
    float li = 1.0f / fmaxf(sqrtf(lss), 1e-12f);
    float ri = 1.0f / fmaxf(sqrtf(rss), 1e-12f);
    av[tid] = l0 * li; av[tid + 256] = l1 * li;
    bv[tid] = r0 * ri; bv[tid + 256] = r1 * ri;
    __syncthreads();

    float s0 = 0.0f, s1 = 0.0f;
#pragma unroll 8
    for (int j = 0; j < 512; ++j) {
      float aj = av[j];
      s0 = fmaf(aj, bv[(j + tid) & 511], s0);
      s1 = fmaf(aj, bv[(j + tid + 256) & 511], s1);
    }
    float css = blockReduceSum(s0 * s0 + s1 * s1, scr);
    float ci = 1.0f / fmaxf(sqrtf(css), 1e-12f);

    float* dst = ph + (size_t)(p - 128) * 512;
    if (t == 1) {
      dst[tid] = araw0[tid];
      dst[tid + 256] = araw0[tid + 256];
    } else if (t == 2) {
      dst[tid] = s0 * ci;
      dst[tid + 256] = s1 * ci;
    }
    __syncthreads();
  }

  for (int i = tid; i < 127 * 512; i += 256)
    phrase_bf[(size_t)b * 127 * 512 + i] = (bf16)ph[i];
}

// ---------------------------------------------------------------------------
// Labels (int32 per JAX default x64-off) -> float tail of d_out.
// ---------------------------------------------------------------------------
__global__ __launch_bounds__(256)
void k_labels(const int* __restrict__ lab, float* __restrict__ outw,
              float* __restrict__ outp) {
  int i = blockIdx.x * blockDim.x + threadIdx.x;  // over 64*255
  if (i >= 64 * 255) return;
  int b = i / 255, col = i % 255;
  float v = (float)lab[i];
  if (col < 128) outw[b * 128 + col] = v;
  else outp[b * 127 + (col - 128)] = v;
}

// ---------------------------------------------------------------------------
// Launcher
// ---------------------------------------------------------------------------
extern "C" void kernel_launch(void* const* d_in, const int* in_sizes, int n_in,
                              void* d_out, int out_size, void* d_ws,
                              size_t ws_size, hipStream_t stream) {
  const float* elmo    = (const float*)d_in[0];
  const float* w_ih_f  = (const float*)d_in[1];
  const float* w_hh_f  = (const float*)d_in[2];
  const float* b_f     = (const float*)d_in[3];
  const float* w_ih_b  = (const float*)d_in[4];
  const float* w_hh_b  = (const float*)d_in[5];
  const float* b_b     = (const float*)d_in[6];
  const float* W1      = (const float*)d_in[7];
  const float* W2      = (const float*)d_in[8];
  const float* word_W  = (const float*)d_in[9];
  const float* word_b  = (const float*)d_in[10];
  const float* phrase_W= (const float*)d_in[11];
  const float* phrase_b= (const float*)d_in[12];
  const int*   comp    = (const int*)d_in[13];
  const int*   pos     = (const int*)d_in[14];
  const int*   lab     = (const int*)d_in[15];

  // Workspace carve (256B aligned)
  size_t off = 0;
  auto carve = [&](size_t bytes) -> void* {
    void* p = (char*)d_ws + off;
    off += (bytes + 255) & ~(size_t)255;
    return p;
  };
  float* xp_f       = (float*)carve((size_t)8192 * 2048 * 4);
  float* xp_b       = (float*)carve((size_t)8192 * 2048 * 4);
  bf16*  x_bf       = (bf16*) carve((size_t)8192 * 1024 * 2);
  bf16*  xrev_bf    = (bf16*) carve((size_t)8192 * 1024 * 2);
  bf16*  wihf_bf    = (bf16*) carve((size_t)2048 * 1024 * 2);
  bf16*  wihb_bf    = (bf16*) carve((size_t)2048 * 1024 * 2);
  bf16*  whhf_bf    = (bf16*) carve((size_t)2048 * 512 * 2);
  bf16*  whhb_bf    = (bf16*) carve((size_t)2048 * 512 * 2);
  bf16*  hs_cat     = (bf16*) carve((size_t)8192 * 1024 * 2);
  bf16*  h_state    = (bf16*) carve((size_t)2 * 64 * 512 * 2);
  float* c_state    = (float*)carve((size_t)2 * 64 * 512 * 4);
  float* gbuf       = (float*)carve((size_t)2 * 64 * 2048 * 4);
  bf16*  Wc         = (bf16*) carve((size_t)512 * 1024 * 2);
  float* comb       = (float*)carve((size_t)8192 * 512 * 4);
  float* wordvec    = (float*)carve((size_t)8192 * 512 * 4);
  bf16*  wordvec_bf = (bf16*) carve((size_t)8192 * 512 * 2);
  bf16*  phrase_bf  = (bf16*) carve((size_t)8128 * 512 * 2);
  bf16*  wordW_bf   = (bf16*) carve((size_t)512 * 512 * 2);
  bf16*  phraseW_bf = (bf16*) carve((size_t)256 * 512 * 2);
  unsigned* cnt     = (unsigned*)carve(256);

  float* out        = (float*)d_out;
  float* out_word   = out;                                  // 8192 x 511
  float* out_phrase = out + (size_t)8192 * 511;             // 8128 x 152
  float* out_wl     = out_phrase + (size_t)8128 * 152;      // 8192
  float* out_pl     = out_wl + 8192;                        // 8128

  // Allow >64KB dynamic LDS (CDNA5 WGP has 320KB).
  (void)hipFuncSetAttribute(reinterpret_cast<const void*>(k_lstm),
                            hipFuncAttributeMaxDynamicSharedMemorySize,
                            2 * 64 * 512 * 2);
  (void)hipFuncSetAttribute(reinterpret_cast<const void*>(k_compose),
                            hipFuncAttributeMaxDynamicSharedMemorySize,
                            127 * 512 * 4);

  // --- Stage 0: pack activations & weights to bf16 (padded where needed) ---
  k_pack<<<8192, 256, 0, stream>>>(elmo, x_bf, 1024, 1024, 1024, 8192);
  k_pack_rev<<<8192, 256, 0, stream>>>(elmo, xrev_bf, 1024);
  k_pack<<<2048, 256, 0, stream>>>(w_ih_f, wihf_bf, 1024, 1024, 1024, 2048);
  k_pack<<<2048, 256, 0, stream>>>(w_ih_b, wihb_bf, 1024, 1024, 1024, 2048);
  k_pack<<<2048, 256, 0, stream>>>(w_hh_f, whhf_bf, 512, 512, 512, 2048);
  k_pack<<<2048, 256, 0, stream>>>(w_hh_b, whhb_bf, 512, 512, 512, 2048);
  k_pack<<<512, 256, 0, stream>>>(W1, Wc, 512, 512, 1024, 512);        // K 0..511
  k_pack<<<512, 256, 0, stream>>>(W2, Wc + 512, 512, 512, 1024, 512);  // K 512..1023
  k_pack<<<512, 256, 0, stream>>>(word_W, wordW_bf, 512, 512, 512, 511);
  k_pack<<<256, 256, 0, stream>>>(phrase_W, phraseW_bf, 512, 512, 512, 152);

  // --- Stage 1: input projections xp = x @ w_ih^T + b  (8192x2048x1024) ---
  k_gemm<<<dim3(16, 128), 256, 0, stream>>>(x_bf, 1024, wihf_bf, 1024, xp_f,
                                            2048, b_f, 1024, 2048, 0);
  k_gemm<<<dim3(16, 128), 256, 0, stream>>>(xrev_bf, 1024, wihb_bf, 1024, xp_b,
                                            2048, b_b, 1024, 2048, 0);

  // --- Stage 2: persistent bidirectional LSTM scan (128 steps in-kernel) ---
  k_zero_u32<<<1, 64, 0, stream>>>(cnt, 64);
  k_lstm<<<64, 256, 2 * 64 * 512 * 2, stream>>>(xp_f, xp_b, whhf_bf, whhb_bf,
                                                h_state, c_state, gbuf, hs_cat,
                                                cnt);

  // --- Stage 3: combined = leaky_relu([fwd|bwd] @ [W1|W2]^T) ---
  k_gemm<<<dim3(4, 128), 256, 0, stream>>>(hs_cat, 1024, Wc, 1024, comb, 512,
                                           nullptr, 1024, 512, 1);

  // --- Stage 4: l2-normalize + scatter via original_pos ---
  k_rownorm<<<8192, 256, 0, stream>>>(comb, pos, wordvec, wordvec_bf);

  // --- Stage 5: 127-step tree composition, phrase table resident in LDS ---
  k_compose<<<64, 256, 127 * 512 * 4, stream>>>(wordvec, comp, phrase_bf);

  // --- Stage 6: output projections ---
  k_gemm<<<dim3(4, 128), 256, 0, stream>>>(wordvec_bf, 512, wordW_bf, 512,
                                           out_word, 511, word_b, 512, 511, 0);
  k_gemm<<<dim3(2, 127), 256, 0, stream>>>(phrase_bf, 512, phraseW_bf, 512,
                                           out_phrase, 152, phrase_b, 512, 152,
                                           0);

  // --- Stage 7: labels ---
  k_labels<<<64, 256, 0, stream>>>(lab, out_wl, out_pl);
}
